// Vision_Transformer_19078244729422
// MI455X (gfx1250) — matrix operand order repaired and checked by reading, not compiled, not run on hardware
//
#include <hip/hip_runtime.h>
#include <hip/hip_bf16.h>

// ---------------------------------------------------------------------------
// ViT-Base forward for MI455X (gfx1250).
// bf16 WMMA everywhere (fp32 accum), TDM (tensor_load_to_lds) tile staging,
// double-buffered k-loop throttled with s_wait_tensorcnt.
// Weights converted once to bf16 pre-transposed [N][K] (~170MB => L2-resident).
// B=32, S=197 (M=6304), D=768, H=12, DH=64, F=3072, L=12, NCLS=1000
// ---------------------------------------------------------------------------

typedef __attribute__((ext_vector_type(16))) __bf16        bf16x16;
typedef __attribute__((ext_vector_type(8)))  float         f32x8;
typedef __attribute__((ext_vector_type(4)))  unsigned int  u32x4;
typedef __attribute__((ext_vector_type(8)))  int           i32x8;
typedef __attribute__((ext_vector_type(4)))  int           i32x4;

union ABfrag {
    bf16x16 v;
    uint4   q[2];
    __bf16  h[16];
};

#define BM 64
#define BN 128
#define BK 32
#define LDA_PAD 40   // bf16 elems per LDS row = 32 + 8 pad (TDM pad: 4 DW per 16 DW)

// ---------------------------------------------------------------------------
// TDM 2D tile load: bf16 tensor, row-major, tile tileCols x tileRows,
// OOB rows/cols (beyond tensorCols/tensorRows from tile start) return zero.
// pad=true inserts 4 DWORDs of LDS padding per 16 DWORDs (row 32 -> 40 bf16).
// Descriptor fields per CDNA5 ISA chapter 8 (D# group0 / group1).
// This toolchain exposes the 6-arg builtin: (g0, g1, g2, g3, g4, cpol).
// ---------------------------------------------------------------------------
__device__ __forceinline__ void tdm_load_2d_bf16(const void* gptr, unsigned ldsOff,
                                                 unsigned tileCols, unsigned tileRows,
                                                 unsigned tensorCols, unsigned tensorRows,
                                                 unsigned strideElems, bool pad)
{
    unsigned long long ga = (unsigned long long)gptr;
    u32x4 g0;
    g0[0] = 1u;                                           // count=1, user mode
    g0[1] = ldsOff;                                       // lds_addr (bytes)
    g0[2] = (unsigned)(ga & 0xFFFFFFFFull);               // global_addr[31:0]
    g0[3] = (unsigned)((ga >> 32) & 0x1FFFFFFull)         // global_addr[56:32]
          | (2u << 30);                                   // type = 2 ("image")
    i32x8 g1;
    unsigned w0 = (1u << 16);                             // data_size = 1 (2 bytes)
    if (pad) w0 |= (1u << 20) | (3u << 22) | (3u << 25);  // pad_en, every 16 DW, +4 DW
    g1[0] = (int)w0;
    g1[1] = (int)((tensorCols & 0xFFFFu) << 16);          // tensor_dim0[15:0] @ bits 63:48
    g1[2] = (int)(((tensorCols >> 16) & 0xFFFFu)          // tensor_dim0[31:16]
          |       ((tensorRows & 0xFFFFu) << 16));        // tensor_dim1[15:0]
    g1[3] = (int)(((tensorRows >> 16) & 0xFFFFu)          // tensor_dim1[31:16]
          |       ((tileCols & 0xFFFFu) << 16));          // tile_dim0
    g1[4] = (int)(tileRows & 0xFFFFu);                    // tile_dim1 (tile_dim2 = 0)
    g1[5] = (int)strideElems;                             // tensor_dim0_stride[31:0]
    g1[6] = 0;                                            // stride hi / dim1_stride lo
    g1[7] = 0;
    i32x4 z4 = {};
    i32x8 z8 = {};
    __builtin_amdgcn_tensor_load_to_lds(g0, g1, z4, z4, z8, 0);
}

// ---------------------------------------------------------------------------
// GEMM: C[M,N] = epi(A[M,K](bf16) * Bt[N,K](bf16)^T + bias [, resid])
// A row-major (lda = K), Bt pre-transposed row-major (ld = K).
// Both tiles staged via TDM with LDS row padding; double-buffered.
// epi: 0 = bias, 1 = bias+GELU(exact), 2 = bias + residual add
// bf16out: store C as bf16 instead of fp32.
// ---------------------------------------------------------------------------
__global__ __launch_bounds__(256, 2)
void vit_gemm_tdm(const __bf16* __restrict__ A, const __bf16* __restrict__ Bt,
                  const float* __restrict__ bias, const float* __restrict__ resid,
                  void* __restrict__ Cout,
                  int M, int N, int K, int ldc, int epi, int bf16out)
{
    __shared__ __bf16 sA[2][BM * LDA_PAD];   // [row][k], k-padded rows
    __shared__ __bf16 sB[2][BN * LDA_PAD];   // [n][k]

    const int tid  = threadIdx.x;
    const int lane = tid & 31;
    const int w    = tid >> 5;
    const int wm   = w & 1;          // 2 wave-rows
    const int wn   = w >> 1;         // 4 wave-cols
    const int rowBase = blockIdx.y * BM;
    const int colBase = blockIdx.x * BN;
    const int nsteps  = K / BK;

    f32x8 acc[2][2] = {};

    if (w == 0) {
        tdm_load_2d_bf16(A + (size_t)rowBase * K, (unsigned)(size_t)&sA[0][0],
                         BK, BM, (unsigned)K, (unsigned)(M - rowBase), (unsigned)K, true);
        tdm_load_2d_bf16(Bt + (size_t)colBase * K, (unsigned)(size_t)&sB[0][0],
                         BK, BN, (unsigned)K, (unsigned)(N - colBase), (unsigned)K, true);
    }

    for (int s = 0; s < nsteps; ++s) {
        const int cur = s & 1;
        if (w == 0) {
            if (s + 1 < nsteps) {
                const int kk = (s + 1) * BK;
                tdm_load_2d_bf16(A + (size_t)rowBase * K + kk, (unsigned)(size_t)&sA[cur ^ 1][0],
                                 BK, BM, (unsigned)(K - kk), (unsigned)(M - rowBase), (unsigned)K, true);
                tdm_load_2d_bf16(Bt + (size_t)colBase * K + kk, (unsigned)(size_t)&sB[cur ^ 1][0],
                                 BK, BN, (unsigned)(K - kk), (unsigned)(N - colBase), (unsigned)K, true);
                __builtin_amdgcn_s_wait_tensorcnt(2);   // oldest pair (cur buffers) done
            } else {
                __builtin_amdgcn_s_wait_tensorcnt(0);
            }
        }
        __syncthreads();

        // ---- fragments (contiguous k: lane half h holds k 16h..16h+15 in A and B) ----
        ABfrag a[2], b[2];
        const int ar  = lane & 15;
    const int akb = (lane >> 4) << 4;   // elems 0-15 = K kb..kb+15
        #pragma unroll
        for (int mi = 0; mi < 2; ++mi) {
            const __bf16* p = &sA[cur][(wm * 32 + mi * 16 + ar) * LDA_PAD + akb];
            a[mi].q[0] = *reinterpret_cast<const uint4*>(p);
            a[mi].q[1] = *reinterpret_cast<const uint4*>(p + 8);
        }
        const int bkb = (lane >> 4) << 4;   // elems 0-15 = K kb2..kb2+15
        #pragma unroll
        for (int ni = 0; ni < 2; ++ni) {
            const __bf16* p = &sB[cur][(wn * 32 + ni * 16 + (lane & 15)) * LDA_PAD + bkb];
            b[ni].q[0] = *reinterpret_cast<const uint4*>(p);
            b[ni].q[1] = *reinterpret_cast<const uint4*>(p + 8);
        }

        #pragma unroll
        for (int mi = 0; mi < 2; ++mi)
            #pragma unroll
            for (int ni = 0; ni < 2; ++ni)
                acc[mi][ni] = __builtin_amdgcn_wmma_f32_16x16x32_bf16(
                    false, a[mi].v, false, b[ni].v, (short)0, acc[mi][ni], false, false);
        __syncthreads();
    }

    // ---- epilogue: C/D layout row = r + 8*(lane>=16), col = lane&15 ----
    const int hs = lane >> 4;
    const int cN = lane & 15;
    #pragma unroll
    for (int mi = 0; mi < 2; ++mi)
        #pragma unroll
        for (int ni = 0; ni < 2; ++ni)
            #pragma unroll
            for (int r = 0; r < 8; ++r) {
                int row = rowBase + wm * 32 + mi * 16 + r + 8 * hs;
                int col = colBase + wn * 32 + ni * 16 + cN;
                if (row < M && col < N) {
                    float v = acc[mi][ni][r] + bias[col];
                    if (epi == 1) {
                        v = 0.5f * v * (1.0f + erff(v * 0.70710678118f));
                    } else if (epi == 2) {
                        v += resid[(size_t)row * ldc + col];
                    }
                    if (bf16out) ((__bf16*)Cout)[(size_t)row * ldc + col] = (__bf16)v;
                    else         ((float*)Cout)[(size_t)row * ldc + col]  = v;
                }
            }
}

// ---------------------------------------------------------------------------
// Fused attention per (b,h): out = softmax(Q K^T / 8) V, all operands bf16.
// K-tile staged by TDM (OOB zero-fill gives the 197->208 padding for free);
// Q fragments loaded straight from global bf16; V^T staged manually.
// ---------------------------------------------------------------------------
__global__ __launch_bounds__(256, 1)
void vit_attention(const __bf16* __restrict__ Q, const __bf16* __restrict__ Km,
                   const __bf16* __restrict__ V, __bf16* __restrict__ O)
{
    __shared__ __bf16 sK[208 * 64];          // [token][dim]   (TDM, rows>=197 zero)
    __shared__ __bf16 sV[64 * 224];          // [dim][token]   (V^T, tokens>=197 zero)
    __shared__ __bf16 sStage[8 * 16 * 40];   // per-wave P re-layout staging

    const int b = blockIdx.y, h = blockIdx.x;
    const int tid = threadIdx.x, lane = tid & 31, w = tid >> 5;
    const __bf16* Qp = Q  + (size_t)b * 197 * 768 + h * 64;
    const __bf16* Kp = Km + (size_t)b * 197 * 768 + h * 64;
    const __bf16* Vp = V  + (size_t)b * 197 * 768 + h * 64;
    __bf16*       Op = O  + (size_t)b * 197 * 768 + h * 64;

    if (w == 0) {
        tdm_load_2d_bf16(Kp, (unsigned)(size_t)&sK[0], 64, 208, 64, 197, 768, false);
    }
    for (int i = tid; i < 64 * 224; i += 256) {
        int d = i / 224, s = i - d * 224;
        sV[i] = (s < 197) ? Vp[(size_t)s * 768 + d] : (__bf16)0.0f;
    }
    if (w == 0) __builtin_amdgcn_s_wait_tensorcnt(0);
    __syncthreads();

    __bf16* myP = &sStage[w * 16 * 40];

    for (int rt = w; rt < 13; rt += 8) {
        const int m0 = rt * 16;
        f32x8 acc[13] = {};

        // ---------------- scores: Q(16x64) x K^T(64x208) ----------------
        const int arow = m0 + (lane & 15);
        #pragma unroll
        for (int kk = 0; kk < 64; kk += 32) {
            ABfrag a;
            const int kb = kk + ((lane >> 4) << 4);
            if (arow < 197) {
                const __bf16* qr = &Qp[(size_t)arow * 768 + kb];
                a.q[0] = *reinterpret_cast<const uint4*>(qr);
                a.q[1] = *reinterpret_cast<const uint4*>(qr + 8);
            } else {
                a.q[0] = make_uint4(0, 0, 0, 0);
                a.q[1] = make_uint4(0, 0, 0, 0);
            }
            const int kb2 = kk + ((lane >> 4) << 4);
            #pragma unroll
            for (int ct = 0; ct < 13; ++ct) {
                ABfrag bb;
                const __bf16* kp = &sK[(ct * 16 + (lane & 15)) * 64 + kb2];
                bb.q[0] = *reinterpret_cast<const uint4*>(kp);
                bb.q[1] = *reinterpret_cast<const uint4*>(kp + 8);
                acc[ct] = __builtin_amdgcn_wmma_f32_16x16x32_bf16(
                    false, a.v, false, bb.v, (short)0, acc[ct], false, false);
            }
        }

        // ---------------- masked softmax (half-wave reductions) ----------------
        #pragma unroll
        for (int r = 0; r < 8; ++r) {
            float mx = -3.0e38f;
            #pragma unroll
            for (int ct = 0; ct < 13; ++ct) {
                float s = acc[ct][r] * 0.125f;
                int col = ct * 16 + (lane & 15);
                if (col >= 197) s = -3.0e38f;
                acc[ct][r] = s;
                mx = fmaxf(mx, s);
            }
            #pragma unroll
            for (int off = 1; off < 16; off <<= 1)
                mx = fmaxf(mx, __shfl_xor(mx, off, 16));
            float sum = 0.0f;
            #pragma unroll
            for (int ct = 0; ct < 13; ++ct) {
                float e = __expf(acc[ct][r] - mx);
                acc[ct][r] = e;
                sum += e;
            }
            #pragma unroll
            for (int off = 1; off < 16; off <<= 1)
                sum += __shfl_xor(sum, off, 16);
            float inv = 1.0f / sum;
            #pragma unroll
            for (int ct = 0; ct < 13; ++ct) acc[ct][r] *= inv;
        }

        // ---------------- out = P(16x224) x V(224x64) ----------------
        f32x8 oacc[4] = {};
        for (int kk2 = 0; kk2 < 224; kk2 += 32) {
            const int ct0 = kk2 >> 4;
            #pragma unroll
            for (int r = 0; r < 8; ++r) {        // C-layout -> A-layout restage
                int prow = r + ((lane >> 4) << 3);
                float v0 = acc[ct0][r];
                float v1 = (ct0 + 1 < 13) ? acc[ct0 + 1][r] : 0.0f;
                myP[prow * 40 + (lane & 15)]      = (__bf16)v0;
                myP[prow * 40 + 16 + (lane & 15)] = (__bf16)v1;
            }
            ABfrag a2;
            const __bf16* pp = &myP[(lane & 15) * 40 + ((lane >> 4) << 4)];
            a2.q[0] = *reinterpret_cast<const uint4*>(pp);
            a2.q[1] = *reinterpret_cast<const uint4*>(pp + 8);
            const int kb2 = kk2 + ((lane >> 4) << 4);
            #pragma unroll
            for (int nt = 0; nt < 4; ++nt) {
                ABfrag bb;
                const __bf16* vp = &sV[(nt * 16 + (lane & 15)) * 224 + kb2];
                bb.q[0] = *reinterpret_cast<const uint4*>(vp);
                bb.q[1] = *reinterpret_cast<const uint4*>(vp + 8);
                oacc[nt] = __builtin_amdgcn_wmma_f32_16x16x32_bf16(
                    false, a2.v, false, bb.v, (short)0, oacc[nt], false, false);
            }
        }
        #pragma unroll
        for (int nt = 0; nt < 4; ++nt)
            #pragma unroll
            for (int r = 0; r < 8; ++r) {
                int srow = m0 + r + 8 * (lane >> 4);
                if (srow < 197)
                    Op[(size_t)srow * 768 + nt * 16 + (lane & 15)] = (__bf16)oacc[nt][r];
            }
    }
}

// ---------------------------------------------------------------------------
// LayerNorm over last dim (768): fp32 in, bf16 out (feeds WMMA GEMMs).
// ---------------------------------------------------------------------------
__global__ void vit_layernorm(const float* __restrict__ X, long ldx,
                              const float* __restrict__ g, const float* __restrict__ bt,
                              __bf16* __restrict__ Y, long ldy, int nrows)
{
    __shared__ float red[256];
    int row = blockIdx.x;
    if (row >= nrows) return;
    const float* xr = X + (size_t)row * (size_t)ldx;
    __bf16*      yr = Y + (size_t)row * (size_t)ldy;
    int tid = threadIdx.x;

    float s = 0.f;
    for (int i = tid; i < 768; i += 256) s += xr[i];
    red[tid] = s; __syncthreads();
    for (int st = 128; st > 0; st >>= 1) { if (tid < st) red[tid] += red[tid + st]; __syncthreads(); }
    float mean = red[0] * (1.0f / 768.0f);
    __syncthreads();

    float v = 0.f;
    for (int i = tid; i < 768; i += 256) { float d = xr[i] - mean; v += d * d; }
    red[tid] = v; __syncthreads();
    for (int st = 128; st > 0; st >>= 1) { if (tid < st) red[tid] += red[tid + st]; __syncthreads(); }
    float rstd = rsqrtf(red[0] * (1.0f / 768.0f) + 1e-5f);

    for (int i = tid; i < 768; i += 256)
        yr[i] = (__bf16)((xr[i] - mean) * rstd * g[i] + bt[i]);
}

// ---------------------------------------------------------------------------
__global__ void vit_im2col(const float* __restrict__ X, __bf16* __restrict__ Ap)
{
    const int total = 6272 * 768;
    for (int idx = blockIdx.x * 256 + threadIdx.x; idx < total; idx += gridDim.x * 256) {
        int m = idx / 768, k = idx - m * 768;
        int b = m / 196,  p = m - b * 196;
        int ph = p / 14,  pw = p - ph * 14;
        int c = k >> 8, rr = (k >> 4) & 15, q = k & 15;
        Ap[idx] = (__bf16)X[(((size_t)(b * 3 + c) * 224 + ph * 16 + rr) * 224) + pw * 16 + q];
    }
}

// dst[N][K] (bf16) = transpose(src[K][N] fp32)
__global__ void vit_wtranspose_bf16(const float* __restrict__ src, __bf16* __restrict__ dst,
                                    int K, int N)
{
    const int total = K * N;
    for (int idx = blockIdx.x * 256 + threadIdx.x; idx < total; idx += gridDim.x * 256) {
        int n = idx / K, k = idx - n * K;
        dst[idx] = (__bf16)src[(size_t)k * N + n];
    }
}

__global__ void vit_cvt_bf16(const float* __restrict__ src, __bf16* __restrict__ dst, int n)
{
    for (int idx = blockIdx.x * 256 + threadIdx.x; idx < n; idx += gridDim.x * 256)
        dst[idx] = (__bf16)src[idx];
}

// x[b,0,:] = cls + pos[0];  x[b,t,:] = patches[b,t-1,:] + pos[t]
__global__ void vit_embed(const float* __restrict__ Pt, const float* __restrict__ cls,
                          const float* __restrict__ pos, float* __restrict__ Xo)
{
    const int total = 6304 * 768;
    for (int idx = blockIdx.x * 256 + threadIdx.x; idx < total; idx += gridDim.x * 256) {
        int row = idx / 768, d = idx - row * 768;
        int b = row / 197, t = row - b * 197;
        float v;
        if (t == 0) v = cls[d] + pos[d];
        else        v = Pt[((size_t)b * 196 + (t - 1)) * 768 + d] + pos[(size_t)t * 768 + d];
        Xo[idx] = v;
    }
}

// ---------------------------------------------------------------------------
static inline void launch_gemm(hipStream_t st, const __bf16* A, const __bf16* Bt,
                               const float* bias, const float* resid, void* C,
                               int M, int N, int K, int ldc, int epi, int bf16out)
{
    dim3 g((N + BN - 1) / BN, (M + BM - 1) / BM);
    vit_gemm_tdm<<<g, 256, 0, st>>>(A, Bt, bias, resid, C, M, N, K, ldc, epi, bf16out);
}

extern "C" void kernel_launch(void* const* d_in, const int* in_sizes, int n_in,
                              void* d_out, int out_size, void* d_ws, size_t ws_size,
                              hipStream_t stream)
{
    (void)in_sizes; (void)n_in; (void)out_size; (void)ws_size;
    const float* X       = (const float*)d_in[0];
    const float* patch_w = (const float*)d_in[1];
    const float* patch_b = (const float*)d_in[2];
    const float* cls_tok = (const float*)d_in[3];
    const float* pos_emb = (const float*)d_in[4];
    const float* ln1_g   = (const float*)d_in[5];
    const float* ln1_b   = (const float*)d_in[6];
    const float* Wq      = (const float*)d_in[7];
    const float* bq      = (const float*)d_in[8];
    const float* Wk      = (const float*)d_in[9];
    const float* bk      = (const float*)d_in[10];
    const float* Wv      = (const float*)d_in[11];
    const float* bv      = (const float*)d_in[12];
    const float* Wo      = (const float*)d_in[13];
    const float* bo      = (const float*)d_in[14];
    const float* ln2_g   = (const float*)d_in[15];
    const float* ln2_b   = (const float*)d_in[16];
    const float* Wf1     = (const float*)d_in[17];
    const float* bf1     = (const float*)d_in[18];
    const float* Wf2     = (const float*)d_in[19];
    const float* bf2     = (const float*)d_in[20];
    const float* lnf_g   = (const float*)d_in[21];
    const float* lnf_b   = (const float*)d_in[22];
    const float* head_w  = (const float*)d_in[23];
    const float* head_b  = (const float*)d_in[24];
    float* out = (float*)d_out;

    const int M = 6304;   // B * S

    // ---- workspace carve (byte offsets, 256B aligned) ----
    char*  wsb = (char*)d_ws;
    size_t off = 0;
    auto alloc = [&](size_t bytes) -> void* {
        void* p = wsb + off;
        off += (bytes + 255) & ~(size_t)255;
        return p;
    };
    float*  x      = (float*) alloc((size_t)M * 768 * 4);       // residual stream (fp32)
    float*  ptmp   = (float*) alloc((size_t)6272 * 768 * 4);    // patch GEMM out (fp32)
    __bf16* hb16   = (__bf16*)alloc((size_t)M * 768 * 2);       // LN out
    __bf16* qb16   = (__bf16*)alloc((size_t)M * 768 * 2);
    __bf16* kb16   = (__bf16*)alloc((size_t)M * 768 * 2);
    __bf16* vb16   = (__bf16*)alloc((size_t)M * 768 * 2);
    __bf16* ab16   = (__bf16*)alloc((size_t)M * 768 * 2);       // attention out
    __bf16* mlp16  = (__bf16*)alloc((size_t)M * 3072 * 2);      // GELU out
    __bf16* pA16   = (__bf16*)alloc((size_t)6272 * 768 * 2);    // im2col
    __bf16* xcls16 = (__bf16*)alloc((size_t)32 * 768 * 2);
    // bf16 weights, pre-transposed to [N][K] (L2-resident working set)
    __bf16* pw16   = (__bf16*)alloc((size_t)768 * 768 * 2);     // patch_w is already [out][k]
    __bf16* wq16   = (__bf16*)alloc((size_t)12 * 768 * 768 * 2);
    __bf16* wk16   = (__bf16*)alloc((size_t)12 * 768 * 768 * 2);
    __bf16* wv16   = (__bf16*)alloc((size_t)12 * 768 * 768 * 2);
    __bf16* wo16   = (__bf16*)alloc((size_t)12 * 768 * 768 * 2);
    __bf16* w116   = (__bf16*)alloc((size_t)12 * 768 * 3072 * 2);
    __bf16* w216   = (__bf16*)alloc((size_t)12 * 768 * 3072 * 2);
    __bf16* head16 = (__bf16*)alloc((size_t)1000 * 768 * 2);

    // ---- one-time weight conversion (bf16, transposed to [N][K]) ----
    vit_cvt_bf16<<<2304, 256, 0, stream>>>(patch_w, pw16, 768 * 768);
    vit_wtranspose_bf16<<<3000, 256, 0, stream>>>(head_w, head16, 768, 1000);
    for (int l = 0; l < 12; ++l) {
        vit_wtranspose_bf16<<<2304, 256, 0, stream>>>(Wq + (size_t)l * 768 * 768, wq16 + (size_t)l * 768 * 768, 768, 768);
        vit_wtranspose_bf16<<<2304, 256, 0, stream>>>(Wk + (size_t)l * 768 * 768, wk16 + (size_t)l * 768 * 768, 768, 768);
        vit_wtranspose_bf16<<<2304, 256, 0, stream>>>(Wv + (size_t)l * 768 * 768, wv16 + (size_t)l * 768 * 768, 768, 768);
        vit_wtranspose_bf16<<<2304, 256, 0, stream>>>(Wo + (size_t)l * 768 * 768, wo16 + (size_t)l * 768 * 768, 768, 768);
        vit_wtranspose_bf16<<<9216, 256, 0, stream>>>(Wf1 + (size_t)l * 768 * 3072, w116 + (size_t)l * 768 * 3072, 768, 3072);
        vit_wtranspose_bf16<<<9216, 256, 0, stream>>>(Wf2 + (size_t)l * 768 * 3072, w216 + (size_t)l * 768 * 3072, 3072, 768);
    }

    // ---- patch embedding ----
    vit_im2col<<<4096, 256, 0, stream>>>(X, pA16);
    launch_gemm(stream, pA16, pw16, patch_b, nullptr, ptmp, 6272, 768, 768, 768, 0, 0);
    vit_embed<<<4096, 256, 0, stream>>>(ptmp, cls_tok, pos_emb, x);

    // ---- transformer blocks ----
    for (int l = 0; l < 12; ++l) {
        const __bf16* wq = wq16 + (size_t)l * 768 * 768;
        const __bf16* wk = wk16 + (size_t)l * 768 * 768;
        const __bf16* wv = wv16 + (size_t)l * 768 * 768;
        const __bf16* wo = wo16 + (size_t)l * 768 * 768;
        const __bf16* w1 = w116 + (size_t)l * 768 * 3072;
        const __bf16* w2 = w216 + (size_t)l * 768 * 3072;

        vit_layernorm<<<M, 256, 0, stream>>>(x, 768, ln1_g + l * 768, ln1_b + l * 768, hb16, 768, M);
        launch_gemm(stream, hb16, wq, bq + l * 768, nullptr, qb16, M, 768, 768, 768, 0, 1);
        launch_gemm(stream, hb16, wk, bk + l * 768, nullptr, kb16, M, 768, 768, 768, 0, 1);
        launch_gemm(stream, hb16, wv, bv + l * 768, nullptr, vb16, M, 768, 768, 768, 0, 1);
        vit_attention<<<dim3(12, 32), 256, 0, stream>>>(qb16, kb16, vb16, ab16);
        launch_gemm(stream, ab16, wo, bo + l * 768, x, x, M, 768, 768, 768, 2, 0);

        vit_layernorm<<<M, 256, 0, stream>>>(x, 768, ln2_g + l * 768, ln2_b + l * 768, hb16, 768, M);
        launch_gemm(stream, hb16, w1, bf1 + l * 3072, nullptr, mlp16, M, 3072, 768, 3072, 1, 1);
        launch_gemm(stream, mlp16, w2, bf2 + l * 768, x, x, M, 768, 3072, 768, 2, 0);
    }

    // ---- final LN on CLS rows + classification head ----
    vit_layernorm<<<32, 256, 0, stream>>>(x, (long)197 * 768, lnf_g, lnf_b, xcls16, 768, 32);
    launch_gemm(stream, xcls16, head16, head_b, nullptr, out, 32, 1000, 768, 1000, 0, 0);
}
